// CursorLiquid_3169685865303
// MI455X (gfx1250) — matrix operand order repaired and checked by reading, not compiled, not run on hardware
//
#include <hip/hip_runtime.h>
#include <hip/hip_bf16.h>

// ---------------------------------------------------------------------------
// Liquid MoE scan for MI455X (gfx1250).
//
// B=512, S=512, D=128, E=32, K=2. Latency-bound sequential scan:
//  - 32 workgroups x 256 threads (8 wave32s); each WG owns 16 batch rows for
//    all 512 steps; h lives in LDS -> no global sync anywhere.
//  - Per step: f32 VALU router + top-2 softmax; (row,k) pairs bucketed by
//    expert; per non-empty expert group a 16x128 output tile is computed as
//    A[16x256](bf16, gathered [x_t|h]) @ B[256x128](bf16, pre-packed
//    [W_in;W_h]) with v_wmma_f32_16x16x32_bf16, f32 accumulate.
//  - A-staging is double-buffered: group i+1's A-tile is built while group i
//    is consumed -> one workgroup barrier per expert group.
//  - Native v_tanh_f32 (TRANS, co-executes with WMMA) for the activation.
//  - Weights pre-packed once per call into d_ws (2 MB, L2 resident) in the
//    exact per-lane order WMMA wants -> inner loop is b128 loads only.
//  - global_prefetch_b8 for next step's x tile.
// ---------------------------------------------------------------------------

#define B_   512
#define S_   512
#define D_   128
#define E_   32
#define DT_F 0.02f

typedef __attribute__((ext_vector_type(16))) __bf16 v16bf;
typedef __attribute__((ext_vector_type(8)))  float  v8f;

#if __has_builtin(__builtin_amdgcn_tanhf)
static __device__ __forceinline__ float fast_tanh(float x) {
  return __builtin_amdgcn_tanhf(x);              // v_tanh_f32 (TRANS)
}
#else
static __device__ __forceinline__ float fast_tanh(float x) { return tanhf(x); }
#endif

// ---------------------------------------------------------------------------
// Pack [W_in ; W_h] (f32, [E][128][128] each) into bf16 WMMA B-tiles.
// Layout: [e][kc(8)][nt(8)][lane(32)][j(16)]  (each lane-block = 16 halfs,
// 32B, contiguous). Lane L holds B[k][n = nt*16+L%16], element j at
// k = kc*32 + (j<8?0:16) + (L/16)*8 + (j&7)  (same k order as the A tile).
// ---------------------------------------------------------------------------
__global__ void liquid_pack_weights(const float* __restrict__ W_in,
                                    const float* __restrict__ W_h,
                                    __bf16* __restrict__ Bpack) {
  int idx = blockIdx.x * blockDim.x + threadIdx.x;   // 32*8*8*32 = 65536
  if (idx >= E_ * 8 * 8 * 32) return;
  int lane = idx & 31;
  int nt   = (idx >> 5) & 7;
  int kc   = (idx >> 8) & 7;
  int e    =  idx >> 11;
  int n    = nt * 16 + (lane & 15);
  int hi   = lane >> 4;
  __bf16* dst = Bpack + (size_t)idx * 16;
#pragma unroll
  for (int j = 0; j < 16; ++j) {
    int k = kc * 32 + ((j < 8) ? 0 : 16) + hi * 8 + (j & 7);  // 0..255 over [Wx;Wh]
    float v = (k < D_)
        ? W_in[((size_t)e * D_ + k) * D_ + n]
        : W_h[((size_t)e * D_ + (k - D_)) * D_ + n];
    dst[j] = (__bf16)v;
  }
}

// ---------------------------------------------------------------------------
// Main persistent-per-tile scan kernel.
// ---------------------------------------------------------------------------
__global__ __launch_bounds__(256, 1)
void liquid_moe_scan(const float* __restrict__ x_seq,
                     const float* __restrict__ Wr,
                     const float* __restrict__ br,
                     const float* __restrict__ b_e,
                     const float* __restrict__ W1,
                     const float* __restrict__ b1,
                     const float* __restrict__ W2,
                     const float* __restrict__ b2,
                     const __bf16* __restrict__ Bpack,
                     float* __restrict__ out) {
  __shared__ float xt[16][D_];                      // x_t tile        (8 KB)
  __shared__ float hb[16][D_];                      // hidden state    (8 KB)
  __shared__ float ub[16][D_];                      // gated mix accum (8 KB)
  __shared__ __align__(16) __bf16 Ast[2][8][32][16]; // A staging x2  (16 KB)
  __shared__ float logit[16][E_];                   // router logits   (2 KB)
  __shared__ float hdn[16][32];                     // head hidden     (2 KB)
  __shared__ int   grp_row[E_][16];
  __shared__ float grp_gate[E_][16];
  __shared__ int   cnt[E_];
  __shared__ int   glist[E_];
  __shared__ int   ngrp;
  __shared__ int   esel[2][16];
  __shared__ float gsel[2][16];

  const int tid  = threadIdx.x;
  const int lane = tid & 31;
  const int wav  = tid >> 5;                        // wave id -> output n-tile
  const int b0   = blockIdx.x * 16;

  // A-tile builder: thread t owns chunk kc=t/32, lane L=t%32 and writes that
  // lane's v16bf slot in the ISA 16-bit A-matrix layout.
  auto buildA = [&](int e, __bf16 (*dst)[32][16]) {
    int c  = cnt[e];
    int kc = tid >> 5, L = tid & 31;
    int m = L & 15, hi = L >> 4;
    int r = (m < c) ? grp_row[e][m] : -1;
#pragma unroll
    for (int j = 0; j < 16; ++j) {
      int v  = j >> 1, hl = j & 1;
      int kl = ((v < 4) ? 0 : 16) + hi * 8 + (v & 3) * 2 + hl;
      int k  = kc * 32 + kl;
      float val = 0.0f;
      if (r >= 0) val = (k < D_) ? xt[r][k] : hb[r][k - D_];
      dst[kc][L][j] = (__bf16)val;
    }
  };

  for (int i = tid; i < 16 * D_; i += 256) (&hb[0][0])[i] = 0.0f;
  __syncthreads();

  for (int t = 0; t < S_; ++t) {
    // ---- load x_t tile (coalesced b128s) + prefetch next step ----
    {
      int r = tid >> 4, c = (tid & 15) * 8;
      const float* src = x_seq + ((size_t)(b0 + r) * S_ + t) * D_ + c;
      float4 a0 = ((const float4*)src)[0];
      float4 a1 = ((const float4*)src)[1];
      *(float4*)&xt[r][c]     = a0;
      *(float4*)&xt[r][c + 4] = a1;
      if (t + 1 < S_) __builtin_prefetch(src + D_, 0, 3);  // global_prefetch_b8
    }
    for (int i = tid; i < 16 * D_; i += 256) (&ub[0][0])[i] = 0.0f;
    if (tid < E_) cnt[tid] = 0;
    __syncthreads();

    // ---- router logits in f32 (expert selection stays in full precision) ----
    {
      int e = tid & 31, r0 = tid >> 5;       // lanes sweep experts: coalesced Wr
      float a0 = br[e], a1 = br[e];
      for (int d = 0; d < D_; ++d) {
        float w = Wr[d * E_ + e];
        a0 += xt[r0][d] * w;
        a1 += xt[r0 + 8][d] * w;
      }
      logit[r0][e] = a0;
      logit[r0 + 8][e] = a1;
    }
    __syncthreads();

    // ---- top-2 + softmax gates per row ----
    if (tid < 16) {
      float v0 = -3.4e38f, v1 = -3.4e38f;
      int i0 = 0, i1 = 0;
      for (int e = 0; e < E_; ++e) {
        float v = logit[tid][e];
        if (v > v0)      { v1 = v0; i1 = i0; v0 = v; i0 = e; }
        else if (v > v1) { v1 = v;  i1 = e; }
      }
      float g1 = __expf(v1 - v0);            // g0 = 1 (v0 is max)
      float inv = 1.0f / (1.0f + g1);
      esel[0][tid] = i0; esel[1][tid] = i1;
      gsel[0][tid] = inv; gsel[1][tid] = g1 * inv;
    }
    __syncthreads();

    // ---- bucket the 32 (row,slot) pairs by expert ----
    if (tid < 32) {
      int r = tid & 15, slot = tid >> 4;
      int e = esel[slot][r];
      int pos = atomicAdd(&cnt[e], 1);       // slot permutation only; result
      grp_row[e][pos]  = r;                  // is order-invariant per pair
      grp_gate[e][pos] = gsel[slot][r];
    }
    if (tid == 0) {                          // deterministic ascending order
      int n = 0;
      for (int e = 0; e < E_; ++e)
        if (cnt[e] > 0) glist[n++] = e;
      ngrp = n;
    }
    __syncthreads();

    // ---- per-expert WMMA tiles, software-pipelined A staging ----
    const int ng = ngrp;
    buildA(glist[0], Ast[0]);
    __syncthreads();
    for (int i = 0; i < ng; ++i) {
      const int e   = glist[i];
      const int c   = cnt[e];
      const int buf = i & 1;

      if (i + 1 < ng) buildA(glist[i + 1], Ast[buf ^ 1]);  // overlap with WMMA

      {
        v8f acc = {0.f, 0.f, 0.f, 0.f, 0.f, 0.f, 0.f, 0.f};
#pragma unroll
        for (int kc = 0; kc < 8; ++kc) {
          v16bf a  = *(const v16bf*)&Ast[buf][kc][lane][0];
          v16bf bm = ((const v16bf*)Bpack)
              [(((size_t)e * 8 + kc) * 8 + wav) * 32 + lane];
          acc = __builtin_amdgcn_wmma_f32_16x16x32_bf16(
              false, a, false, bm, (short)0, acc, false, false);
        }
        int n  = wav * 16 + (lane & 15);
        int hi = lane >> 4;
        float be = b_e[e * D_ + n];
#pragma unroll
        for (int v = 0; v < 8; ++v) {
          int m = v + hi * 8;                // D layout: lane%16 = n, vgpr->m
          if (m < c) {
            int r   = grp_row[e][m];
            float g = grp_gate[e][m];
            ub[r][n] += g * fast_tanh(acc[v] + be);  // unique (r,n) in group
          }
        }
      }
      __syncthreads();                       // next A buffer ready; order u RMW
    }

    // ---- liquid Euler update ----
    for (int i = tid; i < 16 * D_; i += 256) {
      float hv = (&hb[0][0])[i];
      (&hb[0][0])[i] = hv + DT_F * ((&ub[0][0])[i] - hv);
    }
    __syncthreads();
  }

  // ---- head: Linear(128,32) -> exact GELU -> Linear(32,10) -> sigmoid ----
  {
    int j = tid & 31, r0 = tid >> 5;
    for (int rr = r0; rr < 16; rr += 8) {
      float a = b1[j];
      for (int d = 0; d < D_; ++d) a += hb[rr][d] * W1[d * 32 + j];
      hdn[rr][j] = 0.5f * a * (1.0f + erff(a * 0.70710678118654752f));
    }
  }
  __syncthreads();
  if (tid < 160) {
    int r = tid / 10, o = tid % 10;
    float p = b2[o];
#pragma unroll
    for (int j = 0; j < 32; ++j) p += hdn[r][j] * W2[j * 10 + o];
    out[(size_t)(b0 + r) * 10 + o] = 1.0f / (1.0f + __expf(-p));
  }
}

// ---------------------------------------------------------------------------
extern "C" void kernel_launch(void* const* d_in, const int* in_sizes, int n_in,
                              void* d_out, int out_size, void* d_ws, size_t ws_size,
                              hipStream_t stream) {
  const float* x_seq = (const float*)d_in[0];
  const float* Wr    = (const float*)d_in[1];
  const float* br    = (const float*)d_in[2];
  const float* W_in  = (const float*)d_in[3];
  const float* W_h   = (const float*)d_in[4];
  const float* b_e   = (const float*)d_in[5];
  const float* W1    = (const float*)d_in[6];
  const float* b1    = (const float*)d_in[7];
  const float* W2    = (const float*)d_in[8];
  const float* b2    = (const float*)d_in[9];
  float* out = (float*)d_out;
  __bf16* Bpack = (__bf16*)d_ws;               // 2 MB, rebuilt every call

  liquid_pack_weights<<<256, 256, 0, stream>>>(W_in, W_h, Bpack);
  liquid_moe_scan<<<B_ / 16, 256, 0, stream>>>(x_seq, Wr, br, b_e,
                                               W1, b1, W2, b2, Bpack, out);
}
